// Head_42219528520286
// MI455X (gfx1250) — matrix operand order repaired and checked by reading, not compiled, not run on hardware
//
#include <hip/hip_runtime.h>
#include <stdint.h>

#define B_  8
#define T_  2048
#define C_  1024
#define HS_ 64

typedef __attribute__((ext_vector_type(16))) __bf16          bf16x16;
typedef __attribute__((ext_vector_type(2)))  __bf16          bf16x2;
typedef __attribute__((ext_vector_type(8)))  float           f32x8;
typedef __attribute__((ext_vector_type(16))) unsigned short  u16x16;
typedef __attribute__((ext_vector_type(8)))  unsigned short  u16x8;

union FragA {
    u16x16 u;
    bf16x16 v;
    unsigned short s[16];
    unsigned int w32[8];
};

__device__ __forceinline__ unsigned short f2bf(float f) {
    union { float f; uint32_t u; } c; c.f = f;
    uint32_t u = c.u;
    u += 0x7FFFu + ((u >> 16) & 1u);   // round-to-nearest-even
    return (unsigned short)(u >> 16);
}

#if defined(__has_builtin)
#if __has_builtin(__builtin_amdgcn_cvt_pk_bf16_f32)
#define HAVE_PK_BF16 1
#endif
#if __has_builtin(__builtin_amdgcn_sched_barrier)
#define SCHED_FENCE() __builtin_amdgcn_sched_barrier(0)
#endif
#endif
#ifndef SCHED_FENCE
#define SCHED_FENCE()
#endif

// pack two floats -> two bf16 in one u32 (RNE)
__device__ __forceinline__ unsigned int pk_bf16(float a, float b) {
#ifdef HAVE_PK_BF16
    bf16x2 p = __builtin_amdgcn_cvt_pk_bf16_f32(a, b);
    union { bf16x2 v; unsigned int u; } c; c.v = p;
    return c.u;
#else
    return (unsigned int)f2bf(a) | ((unsigned int)f2bf(b) << 16);
#endif
}

// ---------------------------------------------------------------------------
// Kernel 1: convert + transpose weights fp32 [C][HS] -> bf16 [HS][C]
// slot 0 = Wq, slot 1 = Wk, slot 2 = Wv
// ---------------------------------------------------------------------------
__global__ void wt_convert(const float* __restrict__ Wk,
                           const float* __restrict__ Wq,
                           const float* __restrict__ Wv,
                           unsigned short* __restrict__ WT) {
    int idx = blockIdx.x * blockDim.x + threadIdx.x;
    int n = 3 * C_ * HS_;
    if (idx >= n) return;
    int w = idx / (C_ * HS_);
    int r = idx % (C_ * HS_);
    int c = r / HS_;
    int h = r % HS_;
    const float* W = (w == 0) ? Wq : (w == 1) ? Wk : Wv;
    WT[(size_t)w * HS_ * C_ + (size_t)h * C_ + c] = f2bf(W[(size_t)c * HS_ + h]);
}

// ---------------------------------------------------------------------------
// Kernel 2: fused QKV projection.  One wave = one 16-row M tile, all 3
// matrices x 4 N-tiles.  B-fragment groups are double-buffered with hard
// scheduling fences so group w+1's 8 b128 loads are in flight while group
// w's 4 WMMAs execute (partial loadcnt waits instead of drain-to-zero).
// Q pre-scaled by 1/sqrt(HS)=0.125.  V stored transposed [B][HS][T].
// ---------------------------------------------------------------------------
__global__ void __launch_bounds__(256)
qkv_proj(const float* __restrict__ x, const unsigned short* __restrict__ WT,
         unsigned short* __restrict__ Qb, unsigned short* __restrict__ Kb,
         unsigned short* __restrict__ Vt) {
    const int wave = threadIdx.x >> 5;
    const int lane = threadIdx.x & 31;
    const int hi = lane >> 4;      // half-wave select
    const int ln = lane & 15;
    const int mtile = blockIdx.x * 8 + wave;       // 0..1023
    const int mrow = mtile * 16;                   // flattened [B*T] row base

    f32x8 acc[3][4];
    #pragma unroll
    for (int w = 0; w < 3; w++)
        #pragma unroll
        for (int t = 0; t < 4; t++)
            #pragma unroll
            for (int r = 0; r < 8; r++) acc[w][t][r] = 0.0f;

    const float* xrow = x + (size_t)(mrow + ln) * C_;
    const unsigned short* wbase = WT + (size_t)ln * C_ + hi * 16;
    const int koff = hi * 8;

    for (int kb = 0; kb < C_; kb += 32) {
        // A fragment: row m=ln, K pattern {hi*16+0..15}
        const float4* xp0 = (const float4*)(xrow + kb + hi * 16);
        const float4* xp1 = (const float4*)(xrow + kb + hi * 16 + 8);
        float4 f0 = xp0[0], f1 = xp0[1], f2 = xp1[0], f3 = xp1[1];
        FragA a;
        a.w32[0] = pk_bf16(f0.x, f0.y); a.w32[1] = pk_bf16(f0.z, f0.w);
        a.w32[2] = pk_bf16(f1.x, f1.y); a.w32[3] = pk_bf16(f1.z, f1.w);
        a.w32[4] = pk_bf16(f2.x, f2.y); a.w32[5] = pk_bf16(f2.z, f2.w);
        a.w32[6] = pk_bf16(f3.x, f3.y); a.w32[7] = pk_bf16(f3.z, f3.w);

        const unsigned short* wk = wbase + kb;
        FragA bfr[2][4];
        // preload group w=0 (4 fragments = 8 x b128)
        #pragma unroll
        for (int t = 0; t < 4; t++)
            bfr[0][t].u = *(const u16x16*)(wk + (size_t)t * 16 * C_);

        #pragma unroll
        for (int w = 0; w < 3; w++) {
            if (w < 2) {   // issue next matrix's 4 fragment loads first
                #pragma unroll
                for (int t = 0; t < 4; t++)
                    bfr[(w + 1) & 1][t].u = *(const u16x16*)(wk +
                        (size_t)(w + 1) * HS_ * C_ + (size_t)t * 16 * C_);
            }
            SCHED_FENCE();   // loads above must be issued before WMMAs below
            #pragma unroll
            for (int t = 0; t < 4; t++)
                acc[w][t] = __builtin_amdgcn_wmma_f32_16x16x32_bf16(
                    false, a.v, false, bfr[w & 1][t].v, (short)0, acc[w][t],
                    false, false);
            SCHED_FENCE();
        }
    }

    const int b = mrow / T_;
    const int tloc = mrow % T_;
    #pragma unroll
    for (int t = 0; t < 4; t++) {
        int h = t * 16 + ln;
        #pragma unroll
        for (int r = 0; r < 8; r++) {
            int row = tloc + r + hi * 8;
            size_t off = ((size_t)b * T_ + row) * HS_ + h;
            Qb[off] = f2bf(acc[0][t][r] * 0.125f);
            Kb[off] = f2bf(acc[1][t][r]);
        }
        // V transposed: lane owns 8 consecutive t positions -> packed b32 stores
        unsigned int* vrow = (unsigned int*)(Vt +
            ((size_t)b * HS_ + h) * T_ + tloc + hi * 8);
        #pragma unroll
        for (int r = 0; r < 8; r += 2)
            vrow[r >> 1] = pk_bf16(acc[2][t][r], acc[2][t][r + 1]);
    }
}

// ---------------------------------------------------------------------------
// Kernel 3: causal flash attention.  One wave = 16 query rows; online
// softmax over KV chunks of 32.  All 4 K + 4 V fragments are loaded and
// fenced before the S WMMAs, so V loads stay in flight across the entire
// softmax VALU section.  P staged through per-wave LDS to reshape C/D
// layout into A-fragment layout.  No block barriers (waves diverge in
// trip count).
// ---------------------------------------------------------------------------
__global__ void __launch_bounds__(128)
flash_attn(const unsigned short* __restrict__ Qb,
           const unsigned short* __restrict__ Kb,
           const unsigned short* __restrict__ Vt,
           float* __restrict__ out) {
    __shared__ unsigned short Pst[4][16 * 32];
    const int wave = threadIdx.x >> 5;
    const int lane = threadIdx.x & 31;
    const int hi = lane >> 4;
    const int ln = lane & 15;
    const int mtile = blockIdx.x * 4 + wave;       // 0..1023
    const int b = (mtile * 16) / T_;
    const int qbase = (mtile * 16) % T_;

    const unsigned short* Qbase = Qb + (size_t)b * T_ * HS_;
    const unsigned short* Kbase = Kb + (size_t)b * T_ * HS_;
    const unsigned short* Vbase = Vt + (size_t)b * HS_ * T_;

    // Persistent Q fragments: 16x64 = two K-steps of 16x32
    FragA qf[2];
    {
        const unsigned short* qrow = Qbase + (size_t)(qbase + ln) * HS_;
        const int ko = hi * 16;
        #pragma unroll
        for (int s = 0; s < 2; s++) {
            *(u16x8*)&qf[s].s[0] = *(const u16x8*)(qrow + s * 32 + ko);
            *(u16x8*)&qf[s].s[8] = *(const u16x8*)(qrow + s * 32 + ko + 8);
        }
    }

    f32x8 acc[4];
    #pragma unroll
    for (int t = 0; t < 4; t++)
        #pragma unroll
        for (int r = 0; r < 8; r++) acc[t][r] = 0.0f;

    float mprev[8], lsum[8];
    #pragma unroll
    for (int r = 0; r < 8; r++) { mprev[r] = -1e30f; lsum[r] = 0.0f; }

    const int nchunk = (qbase + 15) / 32 + 1;

    for (int c = 0; c < nchunk; c++) {
        const int cb = c * 32;
        if (c + 1 < nchunk)   // hint next K chunk into cache
            __builtin_prefetch(Kbase + (size_t)(cb + 32 + ln) * HS_, 0, 3);

        // Load all chunk operands up front: 4 K fragments + 4 V fragments.
        FragA kf[2][2], vf[4];
        #pragma unroll
        for (int jj = 0; jj < 2; jj++) {
            const unsigned short* krow =
                Kbase + (size_t)(cb + jj * 16 + ln) * HS_ + hi * 16;
            #pragma unroll
            for (int st = 0; st < 2; st++)
                kf[jj][st].u = *(const u16x16*)(krow + st * 32);
        }
        #pragma unroll
        for (int t = 0; t < 4; t++)
            vf[t].u = *(const u16x16*)(Vbase +
                (size_t)(t * 16 + ln) * T_ + cb + hi * 16);
        SCHED_FENCE();   // all 16 b128 loads issued before any WMMA

        // S tile 16x32 = two 16x16 WMMA outputs, K=64 over head dim
        f32x8 s[2];
        #pragma unroll
        for (int jj = 0; jj < 2; jj++) {
            #pragma unroll
            for (int r = 0; r < 8; r++) s[jj][r] = 0.0f;
            #pragma unroll
            for (int st = 0; st < 2; st++)
                s[jj] = __builtin_amdgcn_wmma_f32_16x16x32_bf16(
                    false, qf[st].v, false, kf[jj][st].v, (short)0, s[jj],
                    false, false);
        }

        // Online softmax: lane holds col=ln (and ln+16), rows r + hi*8.
        float pnew[2][8], scl[8];
        #pragma unroll
        for (int r = 0; r < 8; r++) {
            const int qi = qbase + r + hi * 8;
            float s0 = (cb + ln      <= qi) ? s[0][r] : -1e30f;
            float s1 = (cb + 16 + ln <= qi) ? s[1][r] : -1e30f;
            float mx = fmaxf(s0, s1);
            #pragma unroll
            for (int d = 1; d < 16; d <<= 1) mx = fmaxf(mx, __shfl_xor(mx, d, 32));
            float mn = fmaxf(mprev[r], mx);
            float sc = __expf(mprev[r] - mn);
            float p0 = __expf(s0 - mn);
            float p1 = __expf(s1 - mn);
            float rs = p0 + p1;
            #pragma unroll
            for (int d = 1; d < 16; d <<= 1) rs += __shfl_xor(rs, d, 32);
            lsum[r] = lsum[r] * sc + rs;
            mprev[r] = mn;
            scl[r] = sc;
            pnew[0][r] = p0; pnew[1][r] = p1;
        }
        #pragma unroll
        for (int t = 0; t < 4; t++)
            #pragma unroll
            for (int r = 0; r < 8; r++) acc[t][r] *= scl[r];

        // Stage P (16x32 bf16) in this wave's LDS slab; reload as A fragment.
        unsigned short* P = Pst[wave];
        #pragma unroll
        for (int jj = 0; jj < 2; jj++)
            #pragma unroll
            for (int r = 0; r < 8; r++)
                P[(r + hi * 8) * 32 + jj * 16 + ln] = f2bf(pnew[jj][r]);

        FragA pf;
        {
            const unsigned short* prow = P + ln * 32 + hi * 16;
            *(u16x8*)&pf.s[0] = *(const u16x8*)(prow);
            *(u16x8*)&pf.s[8] = *(const u16x8*)(prow + 8);
        }

        // acc += P (16x32) x V (32x16) for each of 4 head tiles
        #pragma unroll
        for (int t = 0; t < 4; t++)
            acc[t] = __builtin_amdgcn_wmma_f32_16x16x32_bf16(
                false, pf.v, false, vf[t].v, (short)0, acc[t], false, false);
    }

    // Epilogue: out = acc / l
    #pragma unroll
    for (int t = 0; t < 4; t++) {
        const int h = t * 16 + ln;
        #pragma unroll
        for (int r = 0; r < 8; r++) {
            const int row = qbase + r + hi * 8;
            out[((size_t)b * T_ + row) * HS_ + h] = acc[t][r] / lsum[r];
        }
    }
}

// ---------------------------------------------------------------------------
extern "C" void kernel_launch(void* const* d_in, const int* in_sizes, int n_in,
                              void* d_out, int out_size, void* d_ws, size_t ws_size,
                              hipStream_t stream) {
    (void)in_sizes; (void)n_in; (void)out_size; (void)ws_size;
    const float* x  = (const float*)d_in[0];
    const float* Wk = (const float*)d_in[1];
    const float* Wq = (const float*)d_in[2];
    const float* Wv = (const float*)d_in[3];
    float* out = (float*)d_out;

    char* ws = (char*)d_ws;
    unsigned short* WT = (unsigned short*)(ws);                              // 384 KB
    unsigned short* Qb = (unsigned short*)(ws + 512 * 1024);                 // 2 MB
    unsigned short* Kb = (unsigned short*)(ws + 512 * 1024 + 2 * 1024 * 1024);
    unsigned short* Vt = (unsigned short*)(ws + 512 * 1024 + 4 * 1024 * 1024);

    const int nW = 3 * C_ * HS_;
    wt_convert<<<(nW + 255) / 256, 256, 0, stream>>>(Wk, Wq, Wv, WT);
    qkv_proj<<<128, 256, 0, stream>>>(x, WT, Qb, Kb, Vt);
    flash_attn<<<256, 128, 0, stream>>>(Qb, Kb, Vt, out);
}
